// RefSelfMultiheadAttn_81681688035575
// MI455X (gfx1250) — matrix operand order repaired and checked by reading, not compiled, not run on hardware
//
#include <hip/hip_runtime.h>
#include <hip/hip_bf16.h>

// Shapes from the reference: S=2048, B=2, E=1024, H=16, HD=64
#define S_LEN 2048
#define BSZ   2
#define EMB   1024
#define NH    16
#define HD    64
#define MROWS (S_LEN * BSZ)   // 4096

typedef __attribute__((ext_vector_type(16))) __bf16 v16bf;
typedef __attribute__((ext_vector_type(8)))  float  v8f;

// ---- CDNA5 async global->LDS path (guarded; falls back to ld/st staging) ----
#if defined(__gfx1250__) && __has_builtin(__builtin_amdgcn_global_load_async_to_lds_b128) && \
    __has_builtin(__builtin_amdgcn_s_wait_asynccnt)
#define USE_ASYNC 1
#else
#define USE_ASYNC 0
#endif

#if USE_ASYNC
typedef int v4i_vs __attribute__((vector_size(16)));
typedef __attribute__((address_space(1))) v4i_vs gas_v4i;   // prints as __device__
typedef __attribute__((address_space(3))) v4i_vs las_v4i;   // LDS
__device__ __forceinline__ void async_copy_b128(const unsigned short* g, unsigned short* l) {
    __builtin_amdgcn_global_load_async_to_lds_b128((gas_v4i*)g, (las_v4i*)l,
                                                   /*offset=*/0, /*cpol=*/0);
}
__device__ __forceinline__ void async_wait0() { __builtin_amdgcn_s_wait_asynccnt(0); }
#endif

union Frag {
    uint4 q[2];
    v16bf v;
};

__device__ __forceinline__ unsigned short f2bf(float f) {
    union { float f; unsigned int u; } c; c.f = f;
    unsigned int lsb = (c.u >> 16) & 1u;
    return (unsigned short)((c.u + 0x7fffu + lsb) >> 16);
}

__device__ __forceinline__ v8f wmma_bf16(const Frag& a, const Frag& b, v8f c) {
    return __builtin_amdgcn_wmma_f32_16x16x32_bf16(
        /*neg_a=*/false, a.v, /*neg_b=*/false, b.v,
        /*c_mod=*/(short)0, c, /*reuse_a=*/false, /*reuse_b=*/false);
}

// ---------------------------------------------------------------- convert
__global__ __launch_bounds__(256) void cvt_f32_bf16(const float* __restrict__ src,
                                                    unsigned short* __restrict__ dst,
                                                    int n) {
    for (int i = blockIdx.x * blockDim.x + threadIdx.x; i < n;
         i += gridDim.x * blockDim.x)
        dst[i] = f2bf(src[i]);
}

// ---------------------------------------------------------------- GEMM  D = X * W^T
// X: [M][K] bf16 row-major, W: [N][K] bf16 row-major.  128x128 tile, BK=64.
// MODE 0: QKV projection -> scatter bf16 into q/k/v [B][NH][S][HD], q scaled
// MODE 1: out projection -> f32 into outF [M][N]
template <int MODE>
__global__ __launch_bounds__(256) void gemm_xwt(const unsigned short* __restrict__ X,
                                                const unsigned short* __restrict__ W,
                                                int M, int N, int K,
                                                float* __restrict__ outF,
                                                unsigned short* __restrict__ qb,
                                                unsigned short* __restrict__ kb,
                                                unsigned short* __restrict__ vb) {
    constexpr int KP = 72;                    // padded pitch: 144B rows, conflict-free
    __shared__ __align__(16) unsigned short Xs[128 * KP];
    __shared__ __align__(16) unsigned short Ws[128 * KP];

    const int tid   = threadIdx.x;
    const int lane  = tid & 31;
    const int w     = tid >> 5;               // 8 waves
    const int wm    = w & 3;                  // 4 row groups of 32
    const int wn    = w >> 2;                 // 2 col groups of 64
    const int nlane = lane & 15;
    const int hl    = lane >> 4;

    const int bn = blockIdx.x;                // N / 128
    const int bm = blockIdx.y;                // M / 128

    v8f acc[2][4];
#pragma unroll
    for (int i = 0; i < 2; ++i)
#pragma unroll
        for (int j = 0; j < 4; ++j) acc[i][j] = (v8f){};

    const int rowL  = tid >> 1;               // 0..127  (2 threads per 128B row)
    const int halfL = tid & 1;

    for (int kt = 0; kt < K; kt += 64) {
        // ---- cooperative tile load: 128 rows x 64 bf16 (64B per thread per tile)
        {
            const unsigned short* gx = X + (size_t)(bm * 128 + rowL) * K + kt + halfL * 32;
            const unsigned short* gw = W + (size_t)(bn * 128 + rowL) * K + kt + halfL * 32;
            unsigned short* lx = &Xs[rowL * KP + halfL * 32];
            unsigned short* lw = &Ws[rowL * KP + halfL * 32];
#if USE_ASYNC
#pragma unroll
            for (int i = 0; i < 4; ++i) async_copy_b128(gx + i * 8, lx + i * 8);
#pragma unroll
            for (int i = 0; i < 4; ++i) async_copy_b128(gw + i * 8, lw + i * 8);
            async_wait0();
#else
#pragma unroll
            for (int i = 0; i < 4; ++i) ((uint4*)lx)[i] = ((const uint4*)gx)[i];
#pragma unroll
            for (int i = 0; i < 4; ++i) ((uint4*)lw)[i] = ((const uint4*)gw)[i];
#endif
        }
        __syncthreads();

        // ---- 16 WMMAs per wave between barriers (2 k-steps of 32)
#pragma unroll
        for (int ks = 0; ks < 64; ks += 32) {
            Frag a[2], b[4];
#pragma unroll
            for (int mi = 0; mi < 2; ++mi) {
                const unsigned short* p = &Xs[(wm * 32 + mi * 16 + nlane) * KP + ks + hl * 8];
                a[mi].q[0] = *(const uint4*)p;
                a[mi].q[1] = *(const uint4*)(p + 16);
            }
#pragma unroll
            for (int ni = 0; ni < 4; ++ni) {
                const unsigned short* p = &Ws[(wn * 64 + ni * 16 + nlane) * KP + ks + hl * 8];
                b[ni].q[0] = *(const uint4*)p;
                b[ni].q[1] = *(const uint4*)(p + 16);
            }
#pragma unroll
            for (int mi = 0; mi < 2; ++mi)
#pragma unroll
                for (int ni = 0; ni < 4; ++ni)
                    acc[mi][ni] = wmma_bf16(a[mi], b[ni], acc[mi][ni]);
        }
        __syncthreads();
    }

    // ---- epilogue
#pragma unroll
    for (int mi = 0; mi < 2; ++mi)
#pragma unroll
        for (int ni = 0; ni < 4; ++ni)
#pragma unroll
            for (int j = 0; j < 8; ++j) {
                int mg = bm * 128 + wm * 32 + mi * 16 + hl * 8 + j;
                int ng = bn * 128 + wn * 64 + ni * 16 + nlane;
                float v = acc[mi][ni][j];
                if (MODE == 0) {
                    int s = mg >> 1, bb = mg & 1;
                    int hh = ng / 192, rem = ng % 192;
                    int which = rem >> 6, d = rem & 63;
                    if (which == 0) v *= 0.125f;  // HD^-0.5 folded into q
                    unsigned short* base = (which == 0) ? qb : (which == 1) ? kb : vb;
                    base[(((size_t)(bb * NH + hh) * S_LEN + s) * HD) + d] = f2bf(v);
                } else {
                    outF[(size_t)mg * N + ng] = v;
                }
            }
}

// ---------------------------------------------------------------- flash attention
// q/k/v: [B][NH][S][HD] bf16 (q pre-scaled). attnO: [S*B][EMB] bf16.
__global__ __launch_bounds__(256) void attn_kernel(const unsigned short* __restrict__ qb,
                                                   const unsigned short* __restrict__ kb,
                                                   const unsigned short* __restrict__ vb,
                                                   unsigned short* __restrict__ attnO) {
    constexpr int TP = 72;                    // padded pitch
    __shared__ __align__(16) unsigned short Ks[64 * TP];       // [t'][d]
    __shared__ __align__(16) unsigned short Vs[64 * TP];       // transposed: [d][t']
    __shared__ __align__(16) unsigned short Ps[8 * 16 * TP];   // per-wave P staging

    const int tid   = threadIdx.x;
    const int lane  = tid & 31;
    const int w     = tid >> 5;
    const int nlane = lane & 15;
    const int hl    = lane >> 4;

    const int bid  = blockIdx.x;              // B*NH*(S/128) = 512
    const int qblk = bid & 15;
    const int h    = (bid >> 4) & 15;
    const int b    = bid >> 8;
    const size_t bh    = (size_t)(b * NH + h) * S_LEN * HD;
    const int    qrow0 = qblk * 128 + w * 16; // this wave's 16 query rows

    // Q fragments (A-layout), loaded once from global
    Frag qf[2];
    {
        const unsigned short* qr = qb + bh + (size_t)(qrow0 + nlane) * HD;
        qf[0].q[0] = *(const uint4*)(qr + hl * 8);
        qf[0].q[1] = *(const uint4*)(qr + 16 + hl * 8);
        qf[1].q[0] = *(const uint4*)(qr + 32 + hl * 8);
        qf[1].q[1] = *(const uint4*)(qr + 48 + hl * 8);
    }

    v8f o[4];
#pragma unroll
    for (int i = 0; i < 4; ++i) o[i] = (v8f){};
    float mrun[8], lrun[8];
#pragma unroll
    for (int j = 0; j < 8; ++j) { mrun[j] = -1e30f; lrun[j] = 0.f; }

    for (int tc = 0; tc < S_LEN; tc += 64) {
        // ---- cooperative chunk load
        {
            int r = tid >> 2, qd = tid & 3;   // K: 64 rows, 4 threads/row
            const unsigned short* gk = kb + bh + (size_t)(tc + r) * HD + qd * 16;
            unsigned short*       lk = &Ks[r * TP + qd * 16];
#if USE_ASYNC
            async_copy_b128(gk, lk);
            async_copy_b128(gk + 8, lk + 8);
#else
            ((uint4*)lk)[0] = ((const uint4*)gk)[0];
            *(uint4*)(lk + 8) = ((const uint4*)gk)[1];
#endif
            int tcol = tid & 63, dg = tid >> 6;  // V transposed into Vs[d][t']
            const uint4* gv = (const uint4*)(vb + bh + (size_t)(tc + tcol) * HD + dg * 16);
            uint4 v0 = gv[0], v1 = gv[1];
            unsigned short tmp[16];
            *(uint4*)&tmp[0] = v0; *(uint4*)&tmp[8] = v1;
#pragma unroll
            for (int i = 0; i < 16; ++i) Vs[(dg * 16 + i) * TP + tcol] = tmp[i];
#if USE_ASYNC
            async_wait0();
#endif
        }
        __syncthreads();

        // ---- scores: S = Q * K^T (q pre-scaled), 16x64 per wave
        v8f sc[4];
#pragma unroll
        for (int nt = 0; nt < 4; ++nt) {
            Frag bk0, bk1;
            const unsigned short* kp = &Ks[(nt * 16 + nlane) * TP];
            bk0.q[0] = *(const uint4*)(kp + hl * 8);
            bk0.q[1] = *(const uint4*)(kp + 16 + hl * 8);
            bk1.q[0] = *(const uint4*)(kp + 32 + hl * 8);
            bk1.q[1] = *(const uint4*)(kp + 48 + hl * 8);
            v8f c = (v8f){};
            c = wmma_bf16(qf[0], bk0, c);
            c = wmma_bf16(qf[1], bk1, c);
            sc[nt] = c;
        }

        // ---- online softmax (row stats live in the owning 16-lane half)
#pragma unroll
        for (int j = 0; j < 8; ++j) {
            float cm = fmaxf(fmaxf(sc[0][j], sc[1][j]), fmaxf(sc[2][j], sc[3][j]));
            cm = fmaxf(cm, __shfl_xor(cm, 1));
            cm = fmaxf(cm, __shfl_xor(cm, 2));
            cm = fmaxf(cm, __shfl_xor(cm, 4));
            cm = fmaxf(cm, __shfl_xor(cm, 8));
            float nm    = fmaxf(mrun[j], cm);
            float alpha = __expf(mrun[j] - nm);
            mrun[j] = nm;
            float rs = 0.f;
#pragma unroll
            for (int nt = 0; nt < 4; ++nt) {
                float p = __expf(sc[nt][j] - nm);
                sc[nt][j] = p;
                rs += p;
            }
            rs += __shfl_xor(rs, 1);
            rs += __shfl_xor(rs, 2);
            rs += __shfl_xor(rs, 4);
            rs += __shfl_xor(rs, 8);
            lrun[j] = lrun[j] * alpha + rs;
#pragma unroll
            for (int nt = 0; nt < 4; ++nt) o[nt][j] *= alpha;
        }

        // ---- restage P through per-wave LDS: C-layout f32 -> A-layout bf16
        unsigned short* pw = &Ps[w * 16 * TP];
#pragma unroll
        for (int nt = 0; nt < 4; ++nt)
#pragma unroll
            for (int j = 0; j < 8; ++j)
                pw[(hl * 8 + j) * TP + nt * 16 + nlane] = f2bf(sc[nt][j]);

        Frag pa[2];
        {
            const unsigned short* pr = pw + nlane * TP;
            pa[0].q[0] = *(const uint4*)(pr + hl * 8);
            pa[0].q[1] = *(const uint4*)(pr + 16 + hl * 8);
            pa[1].q[0] = *(const uint4*)(pr + 32 + hl * 8);
            pa[1].q[1] = *(const uint4*)(pr + 48 + hl * 8);
        }

        // ---- O += P * V  (Vs is [d][t'], so the B-loader reads rows n=d)
#pragma unroll
        for (int nt = 0; nt < 4; ++nt) {
            Frag bv0, bv1;
            const unsigned short* vp = &Vs[(nt * 16 + nlane) * TP];
            bv0.q[0] = *(const uint4*)(vp + hl * 8);
            bv0.q[1] = *(const uint4*)(vp + 16 + hl * 8);
            bv1.q[0] = *(const uint4*)(vp + 32 + hl * 8);
            bv1.q[1] = *(const uint4*)(vp + 48 + hl * 8);
            o[nt] = wmma_bf16(pa[0], bv0, o[nt]);
            o[nt] = wmma_bf16(pa[1], bv1, o[nt]);
        }
        __syncthreads();
    }

    // ---- epilogue: normalize, write attn rows as bf16 [s*B+b][h*64+d]
#pragma unroll
    for (int nt = 0; nt < 4; ++nt)
#pragma unroll
        for (int j = 0; j < 8; ++j) {
            int s = qrow0 + hl * 8 + j;
            int e = h * HD + nt * 16 + nlane;
            float val = o[nt][j] / lrun[j];
            attnO[(size_t)(s * BSZ + b) * EMB + e] = f2bf(val);
        }
}

// ---------------------------------------------------------------- launch
extern "C" void kernel_launch(void* const* d_in, const int* in_sizes, int n_in,
                              void* d_out, int out_size, void* d_ws, size_t ws_size,
                              hipStream_t stream) {
    const float* query  = (const float*)d_in[0];
    // d_in[1] (key) and d_in[2] (value) are unused by the reference math.
    const float* w_in   = (const float*)d_in[3];   // [3E][E]
    const float* w_out  = (const float*)d_in[4];   // [E][E]
    float*       out    = (float*)d_out;

    char* ws = (char*)d_ws;
    unsigned short* Xq   = (unsigned short*)(ws);                       // 4096x1024
    unsigned short* Win  = (unsigned short*)(ws + (8u << 20));          // 3072x1024
    unsigned short* Wout = (unsigned short*)(ws + (14u << 20));         // 1024x1024
    unsigned short* qb   = (unsigned short*)(ws + (16u << 20));         // [B][H][S][HD]
    unsigned short* kb   = (unsigned short*)(ws + (24u << 20));
    unsigned short* vb   = (unsigned short*)(ws + (32u << 20));
    unsigned short* attn = (unsigned short*)(ws + (40u << 20));         // 4096x1024

    cvt_f32_bf16<<<2048, 256, 0, stream>>>(query, Xq,  MROWS * EMB);
    cvt_f32_bf16<<<2048, 256, 0, stream>>>(w_in,  Win, 3 * EMB * EMB);
    cvt_f32_bf16<<<1024, 256, 0, stream>>>(w_out, Wout, EMB * EMB);

    // QKV projection: [4096x1024] x [3072x1024]^T, scatter into q/k/v
    gemm_xwt<0><<<dim3(24, 32), 256, 0, stream>>>(Xq, Win, MROWS, 3 * EMB, EMB,
                                                  nullptr, qb, kb, vb);
    // fused flash attention
    attn_kernel<<<BSZ * NH * (S_LEN / 128), 256, 0, stream>>>(qb, kb, vb, attn);
    // output projection: [4096x1024] x [1024x1024]^T -> f32 out
    gemm_xwt<1><<<dim3(8, 32), 256, 0, stream>>>(attn, Wout, MROWS, EMB, EMB,
                                                 out, nullptr, nullptr, nullptr);
}
